// GrownVideoTransformer_81621558493679
// MI455X (gfx1250) — matrix operand order repaired and checked by reading, not compiled, not run on hardware
//
#include <hip/hip_runtime.h>

// ---------------------------------------------------------------------------
// Problem constants (match reference)
// ---------------------------------------------------------------------------
#define BZ   4
#define TT   16
#define IMGD 256
#define PCH  16
#define PP_  256   // patches per frame (G*G)
#define PQ   256   // pixels per patch (PATCH*PATCH)
#define HH   512
#define FFD  2048
#define LL   4
#define NHD  8
#define HD   64
#define NTOK (BZ*TT*PP_)   // 16384
#define BT   (BZ*TT)       // 64

typedef __bf16 bf16;
typedef __attribute__((ext_vector_type(16))) __bf16 v16bf;
typedef __attribute__((ext_vector_type(8)))  __bf16 v8bf;
typedef __attribute__((ext_vector_type(8)))  float  v8f;

union V16U { v16bf v; v8bf h[2]; };

__device__ __forceinline__ bf16 f2bf(float x) {
  union { float f; unsigned u; } a; a.f = x;
  unsigned r = a.u + 0x7fffu + ((a.u >> 16) & 1u);       // round-to-nearest-even
  union { unsigned short s; bf16 b; } o; o.s = (unsigned short)(r >> 16);
  return o.b;
}
__device__ __forceinline__ float bf2f(bf16 x) {
  union { unsigned short s; bf16 b; } i; i.b = x;
  union { unsigned u; float f; } o; o.u = ((unsigned)i.s) << 16;
  return o.f;
}

// ---------------------------------------------------------------------------
// Generic batched WMMA GEMM:  C[z][M][N] = A[z][M][K] * Bt[z][N][K]^T
// Bt is "weights as stored": row n holds the K axis contiguously, matching the
// CDNA5 16-bit B-fragment layout (lanes 0-15: K {0..7,16..23}, lanes 16-31: K {8..15,24..31}).
// Wave tile = (16*MR) x (16*NSUB), both exact divisors of M,N at every call
// site, so there are NO branches inside the K loop (EXEC stays all-ones).
// Batch index z decomposes as (i2, i1, i0) with per-operand strides.
// ---------------------------------------------------------------------------
struct GP {
  const bf16* __restrict__ A;
  const bf16* __restrict__ Bt;
  void* __restrict__ C;
  const float* __restrict__ bias;
  const float* __restrict__ res;
  const float* __restrict__ mask;
  int M, N, K;
  int lda, ldb, ldc, mld;
  float scale;
  int c1, c0;
  long long sA2, sA1, sA0, sB2, sB1, sB0, sC2, sC1, sC0;
};

template<int OBF, int BIAS, int RELU, int RES, int MASKB, int MR, int NSUB>
__global__ __launch_bounds__(128)
void wmma_gemm(GP p) {
  const int lane   = threadIdx.x & 31;
  const int tilesN = p.N / (16 * NSUB);
  const int tilesM = p.M / (16 * MR);
  const int wid    = blockIdx.x * 4 + (threadIdx.x >> 5);
  if (wid >= tilesM * tilesN) return;                 // wave-uniform
  const int m0 = (wid / tilesN) * (16 * MR);
  const int n0 = (wid % tilesN) * (16 * NSUB);

  const int z  = blockIdx.z;
  const int i0 = z % p.c0;
  const int zz = z / p.c0;
  const int i1 = zz % p.c1;
  const int i2 = zz / p.c1;
  const bf16* A  = p.A  + (long long)i2*p.sA2 + (long long)i1*p.sA1 + (long long)i0*p.sA0;
  const bf16* Bt = p.Bt + (long long)i2*p.sB2 + (long long)i1*p.sB1 + (long long)i0*p.sB0;
  const long long coff = (long long)i2*p.sC2 + (long long)i1*p.sC1 + (long long)i0*p.sC0;

  v8f acc[MR][NSUB];
#pragma unroll
  for (int i = 0; i < MR; ++i)
#pragma unroll
    for (int j = 0; j < NSUB; ++j)
#pragma unroll
      for (int r = 0; r < 8; ++r) acc[i][j][r] = 0.0f;

  // A fragment (16x32 bf16): lane<16 -> row=lane,    K chunks {0..7, 16..23}
  //                          lane>=16 -> row=lane-16, K chunks {8..15, 24..31}
  const int arow = m0 + (lane & 15);
  const int aks  = (lane >> 4) << 3;   // 0 or 8
  // B fragment (32x16 bf16): lanes 0-15 -> K {0..7, 16..23}, lanes 16-31 -> K {8..15, 24..31}
  const int bks  = (lane >> 4) << 3;   // 0 or 8
  const int bcl  = lane & 15;

  for (int k0 = 0; k0 < p.K; k0 += 32) {
    V16U a[MR];
#pragma unroll
    for (int i = 0; i < MR; ++i) {
      const bf16* ap = A + (long long)(arow + i * 16) * p.lda + (k0 + aks);
      a[i].h[0] = *(const v8bf*)(ap);
      a[i].h[1] = *(const v8bf*)(ap + 16);
      if (k0 + 32 < p.K) __builtin_prefetch(ap + 32, 0, 3);   // global_prefetch
    }
#pragma unroll
    for (int j = 0; j < NSUB; ++j) {
      V16U b;
      const bf16* bp = Bt + (long long)(n0 + j * 16 + bcl) * p.ldb + (k0 + bks);
      b.h[0] = *(const v8bf*)(bp);
      b.h[1] = *(const v8bf*)(bp + 16);
#pragma unroll
      for (int i = 0; i < MR; ++i)
        acc[i][j] = __builtin_amdgcn_wmma_f32_16x16x32_bf16(
            false, a[i].v, false, b.v, (short)0, acc[i][j], false, false);
    }
  }

  // C layout: lane<16 -> col=lane, rows +0..7 ; lane>=16 -> col=lane-16, rows +8..15
  const int ccl = lane & 15;
#pragma unroll
  for (int i = 0; i < MR; ++i) {
    const int crow0 = m0 + i * 16 + ((lane >> 4) << 3);
#pragma unroll
    for (int j = 0; j < NSUB; ++j) {
      const int col = n0 + j * 16 + ccl;
      float bvadd = BIAS ? p.bias[col] : 0.0f;
#pragma unroll
      for (int r = 0; r < 8; ++r) {
        const int row = crow0 + r;
        float v = acc[i][j][r];
        if (MASKB) v = v * p.scale + p.mask[row * p.mld + col] * 20.0f - 10.0f;
        if (BIAS)  v += bvadd;
        const long long idx = coff + (long long)row * p.ldc + col;
        if (RES)   v += p.res[idx];
        if (RELU)  v = v > 0.0f ? v : 0.0f;
        if (OBF) ((bf16*)p.C)[idx] = f2bf(v);
        else     ((float*)p.C)[idx] = v;
      }
    }
  }
}

template<int OBF, int BIAS, int RELU, int RES, int MASKB, int MR, int NSUB>
static void gemm_launch(hipStream_t s, const GP& p, int nbatch) {
  int tiles = (p.M / (16 * MR)) * (p.N / (16 * NSUB));
  dim3 grid((tiles + 3) / 4, 1, nbatch);
  wmma_gemm<OBF, BIAS, RELU, RES, MASKB, MR, NSUB><<<grid, dim3(128), 0, s>>>(p);
}

static GP gp(const bf16* A, int lda, const bf16* Bt, int ldb, void* C, int ldc,
             int M, int N, int K) {
  GP p{};
  p.A = A; p.Bt = Bt; p.C = C;
  p.lda = lda; p.ldb = ldb; p.ldc = ldc;
  p.M = M; p.N = N; p.K = K;
  p.c1 = 1; p.c0 = 1; p.scale = 1.0f; p.mld = 1;
  return p;
}

// ---------------------------------------------------------------------------
// Elementwise / reduction kernels
// ---------------------------------------------------------------------------
__global__ void k_conv_bf16(bf16* dst, const float* src, long long n) {
  long long i = (long long)blockIdx.x * 256 + threadIdx.x;
  if (i < n) dst[i] = f2bf(src[i]);
}
__global__ void k_conv_bf16_mask(bf16* dst, const float* src, const float* m, long long n) {
  long long i = (long long)blockIdx.x * 256 + threadIdx.x;
  if (i < n) dst[i] = f2bf(src[i] * m[i]);
}
__global__ void k_extract_patches(bf16* xp, const float* x) {
  long long i = (long long)blockIdx.x * 256 + threadIdx.x;
  if (i >= (long long)NTOK * PQ) return;
  int q  = (int)(i & 255);
  int pi = (int)((i >> 8) & 255);
  int bt = (int)(i >> 16);
  int gy = pi >> 4, gx = pi & 15, py = q >> 4, px = q & 15;
  float v = x[((long long)bt * IMGD + gy * PCH + py) * IMGD + gx * PCH + px];
  xp[i] = f2bf(v);
}
__global__ void k_add_pos(float* h, const float* tpos, const float* spos) {
  long long i = (long long)blockIdx.x * 256 + threadIdx.x;
  if (i >= (long long)NTOK * HH) return;
  int hh = (int)(i & 511);
  int pi = (int)((i >> 9) & 255);
  int t  = (int)((i >> 17) & 15);
  h[i] += tpos[t * HH + hh] + spos[pi * HH + hh];
}
__global__ __launch_bounds__(256)
void k_layernorm(bf16* y, const float* x, const float* g, const float* b) {
  const int row = blockIdx.x;
  const float* xr = x + (long long)row * HH;
  float s = 0.f, s2 = 0.f;
  for (int i = threadIdx.x; i < HH; i += 256) { float v = xr[i]; s += v; s2 += v * v; }
#pragma unroll
  for (int o = 16; o; o >>= 1) { s += __shfl_xor(s, o, 32); s2 += __shfl_xor(s2, o, 32); }
  __shared__ float sh[16];
  const int w = threadIdx.x >> 5;
  if ((threadIdx.x & 31) == 0) { sh[w] = s; sh[w + 8] = s2; }
  __syncthreads();
  if (threadIdx.x == 0) {
    float t = 0.f, t2 = 0.f;
    for (int i = 0; i < 8; ++i) { t += sh[i]; t2 += sh[i + 8]; }
    sh[0] = t; sh[8] = t2;
  }
  __syncthreads();
  float mean = sh[0] / (float)HH;
  float var  = sh[8] / (float)HH - mean * mean;
  float rstd = rsqrtf(var + 1e-5f);
  bf16* yr = y + (long long)row * HH;
  for (int i = threadIdx.x; i < HH; i += 256)
    yr[i] = f2bf((xr[i] - mean) * rstd * g[i] + b[i]);
}
// V^T for spatial attention: vt[bt][e][s] = v[bt][s][e]   (bf16)
__global__ void k_vt_spatial(bf16* vt, const float* v) {
  long long i = (long long)blockIdx.x * 256 + threadIdx.x;
  if (i >= (long long)BT * HH * PP_) return;
  int s  = (int)(i & 255);
  int e  = (int)((i >> 8) & 511);
  int bt = (int)(i >> 17);
  vt[i] = f2bf(v[((long long)bt * PP_ + s) * HH + e]);
}
// V^T for temporal attention, K padded to 32: vt[(b*P+p)][e][t] (t in 0..31)
__global__ void k_vt_temporal(bf16* vt, const float* v) {
  long long i = (long long)blockIdx.x * 256 + threadIdx.x;
  if (i >= (long long)BZ * PP_ * HH * 32) return;
  int t  = (int)(i & 31);
  int e  = (int)((i >> 5) & 511);
  int bp = (int)(i >> 14);
  int b = bp >> 8, pp = bp & 255;
  vt[i] = (t < TT) ? f2bf(v[(((long long)(b * TT + t)) * PP_ + pp) * HH + e])
                   : f2bf(0.0f);
}
__global__ __launch_bounds__(128)
void k_softmax256(bf16* d, long long nrows) {
  const long long row = (long long)blockIdx.x * 4 + (threadIdx.x >> 5);
  if (row >= nrows) return;
  const int lane = threadIdx.x & 31;
  bf16* r = d + row * 256;
  float v[8];
  float mx = -1e30f;
#pragma unroll
  for (int i = 0; i < 8; ++i) { v[i] = bf2f(r[i * 32 + lane]); mx = fmaxf(mx, v[i]); }
#pragma unroll
  for (int o = 16; o; o >>= 1) mx = fmaxf(mx, __shfl_xor(mx, o, 32));
  float s = 0.f;
#pragma unroll
  for (int i = 0; i < 8; ++i) { v[i] = __expf(v[i] - mx); s += v[i]; }
#pragma unroll
  for (int o = 16; o; o >>= 1) s += __shfl_xor(s, o, 32);
  float inv = 1.0f / s;
#pragma unroll
  for (int i = 0; i < 8; ++i) r[i * 32 + lane] = f2bf(v[i] * inv);
}
// rows of 16 stored with stride 32; zero-pads cols 16..31 for the K=32 GEMM
__global__ void k_softmax16(bf16* d, long long nrows) {
  long long row = (long long)blockIdx.x * 256 + threadIdx.x;
  if (row >= nrows) return;
  bf16* r = d + row * 32;
  float v[16];
  float mx = -1e30f;
#pragma unroll
  for (int i = 0; i < 16; ++i) { v[i] = bf2f(r[i]); mx = fmaxf(mx, v[i]); }
  float s = 0.f;
#pragma unroll
  for (int i = 0; i < 16; ++i) { v[i] = __expf(v[i] - mx); s += v[i]; }
  float inv = 1.0f / s;
#pragma unroll
  for (int i = 0; i < 16; ++i) { r[i] = f2bf(v[i] * inv); r[16 + i] = f2bf(0.0f); }
}
__global__ void k_unpatch_sigmoid(float* out, const float* patches) {
  long long i = (long long)blockIdx.x * 256 + threadIdx.x;
  if (i >= (long long)BZ * IMGD * IMGD) return;
  int xc = (int)(i & 255);
  int yc = (int)((i >> 8) & 255);
  int b  = (int)(i >> 16);
  int gy = yc >> 4, py = yc & 15, gx = xc >> 4, px = xc & 15;
  float v = patches[((long long)(b * PP_) + gy * 16 + gx) * PQ + py * 16 + px];
  out[i] = 1.0f / (1.0f + __expf(-v));
}

// ---------------------------------------------------------------------------
// Host orchestration
// ---------------------------------------------------------------------------
#define LAUNCH_N(kern, n, ...) \
  kern<<<dim3((unsigned)(((n) + 255) / 256)), dim3(256), 0, stream>>>(__VA_ARGS__)

extern "C" void kernel_launch(void* const* d_in, const int* in_sizes, int n_in,
                              void* d_out, int out_size, void* d_ws, size_t ws_size,
                              hipStream_t stream) {
  (void)in_sizes; (void)n_in; (void)out_size; (void)ws_size;

  const float* X         = (const float*)d_in[0];
  const float* S_MASK    = (const float*)d_in[1];
  const float* T_MASK    = (const float*)d_in[2];
  const float* WO_MASKS  = (const float*)d_in[3];
  const float* FF_MASKS  = (const float*)d_in[4];
  const float* SKIP_MASKS= (const float*)d_in[5];
  const float* HEAD_MASK = (const float*)d_in[6];
  const float* PATCH_W   = (const float*)d_in[7];
  const float* PATCH_B   = (const float*)d_in[8];
  const float* TPOS      = (const float*)d_in[9];
  const float* SPOS      = (const float*)d_in[10];
  const float* SQ_W = (const float*)d_in[11]; const float* SQ_B = (const float*)d_in[12];
  const float* SK_W = (const float*)d_in[13]; const float* SK_B = (const float*)d_in[14];
  const float* SV_W = (const float*)d_in[15]; const float* SV_B = (const float*)d_in[16];
  const float* TQ_W = (const float*)d_in[17]; const float* TQ_B = (const float*)d_in[18];
  const float* TK_W = (const float*)d_in[19]; const float* TK_B = (const float*)d_in[20];
  const float* TV_W = (const float*)d_in[21]; const float* TV_B = (const float*)d_in[22];
  const float* SO_W = (const float*)d_in[23]; const float* TO_W = (const float*)d_in[24];
  const float* LN_S_G = (const float*)d_in[25]; const float* LN_S_B = (const float*)d_in[26];
  const float* LN_T_G = (const float*)d_in[27]; const float* LN_T_B = (const float*)d_in[28];
  const float* LN_F_G = (const float*)d_in[29]; const float* LN_F_B = (const float*)d_in[30];
  const float* FF1_W = (const float*)d_in[31]; const float* FF1_B = (const float*)d_in[32];
  const float* FF2_W = (const float*)d_in[33]; const float* FF2_B = (const float*)d_in[34];
  const float* SKIP_W = (const float*)d_in[35];
  const float* LNF_G = (const float*)d_in[36]; const float* LNF_B = (const float*)d_in[37];
  const float* HEAD_W = (const float*)d_in[38]; const float* HEAD_B = (const float*)d_in[39];

  // ---- workspace carve-out (256B aligned) ----
  char* ws = (char*)d_ws;
  long long off = 0;
  auto alloc = [&](long long bytes) -> char* {
    char* p = ws + off;
    off = (off + bytes + 255) & ~255LL;
    return p;
  };
  bf16* w_patch = (bf16*)alloc((long long)HH * PQ * 2);
  bf16* w_sq = (bf16*)alloc((long long)LL * HH * HH * 2);
  bf16* w_sk = (bf16*)alloc((long long)LL * HH * HH * 2);
  bf16* w_sv = (bf16*)alloc((long long)LL * HH * HH * 2);
  bf16* w_tq = (bf16*)alloc((long long)LL * HH * HH * 2);
  bf16* w_tk = (bf16*)alloc((long long)LL * HH * HH * 2);
  bf16* w_tv = (bf16*)alloc((long long)LL * HH * HH * 2);
  bf16* w_so = (bf16*)alloc((long long)LL * HH * HH * 2);
  bf16* w_to = (bf16*)alloc((long long)LL * HH * HH * 2);
  bf16* w_ff1 = (bf16*)alloc((long long)LL * FFD * HH * 2);
  bf16* w_ff2 = (bf16*)alloc((long long)LL * HH * FFD * 2);
  bf16* w_skip = (bf16*)alloc((long long)6 * HH * HH * 2);
  bf16* w_head = (bf16*)alloc((long long)PQ * HH * 2);

  bf16*  xp_bf  = (bf16*)alloc((long long)NTOK * PQ * 2);
  float* hbuf   = (float*)alloc((long long)NTOK * HH * 4);
  bf16*  ln_bf  = (bf16*)alloc((long long)NTOK * HH * 2);
  bf16*  q_bf   = (bf16*)alloc((long long)NTOK * HH * 2);
  bf16*  k_bf   = (bf16*)alloc((long long)NTOK * HH * 2);
  float* vbuf   = (float*)alloc((long long)NTOK * HH * 4);
  bf16*  vt_bf  = (bf16*)alloc((long long)BZ * PP_ * HH * 32 * 2);  // max of both uses
  bf16*  sc_bf  = (bf16*)alloc((long long)BT * NHD * PP_ * PP_ * 2);
  bf16*  o_bf   = (bf16*)alloc((long long)NTOK * HH * 2);
  bf16*  ff1_bf = (bf16*)alloc((long long)NTOK * FFD * 2);
  bf16*  act_bf[3];
  for (int i = 0; i < 3; ++i) act_bf[i] = (bf16*)alloc((long long)NTOK * HH * 2);
  float* patches = (float*)alloc((long long)BZ * PP_ * PQ * 4);

  // ---- weight conversion (mask folded in) ----
  LAUNCH_N(k_conv_bf16, (long long)HH * PQ, w_patch, PATCH_W, (long long)HH * PQ);
  for (int l = 0; l < LL; ++l) {
    long long oHH = (long long)l * HH * HH;
    long long oFH = (long long)l * FFD * HH;
    LAUNCH_N(k_conv_bf16, (long long)HH * HH, w_sq + oHH, SQ_W + oHH, (long long)HH * HH);
    LAUNCH_N(k_conv_bf16, (long long)HH * HH, w_sk + oHH, SK_W + oHH, (long long)HH * HH);
    LAUNCH_N(k_conv_bf16, (long long)HH * HH, w_sv + oHH, SV_W + oHH, (long long)HH * HH);
    LAUNCH_N(k_conv_bf16, (long long)HH * HH, w_tq + oHH, TQ_W + oHH, (long long)HH * HH);
    LAUNCH_N(k_conv_bf16, (long long)HH * HH, w_tk + oHH, TK_W + oHH, (long long)HH * HH);
    LAUNCH_N(k_conv_bf16, (long long)HH * HH, w_tv + oHH, TV_W + oHH, (long long)HH * HH);
    LAUNCH_N(k_conv_bf16_mask, (long long)HH * HH, w_so + oHH, SO_W + oHH, WO_MASKS + oHH, (long long)HH * HH);
    LAUNCH_N(k_conv_bf16_mask, (long long)HH * HH, w_to + oHH, TO_W + oHH, WO_MASKS + oHH, (long long)HH * HH);
    LAUNCH_N(k_conv_bf16_mask, (long long)FFD * HH, w_ff1 + oFH, FF1_W + oFH, FF_MASKS + oFH, (long long)FFD * HH);
    LAUNCH_N(k_conv_bf16, (long long)HH * FFD, w_ff2 + oFH, FF2_W + oFH, (long long)HH * FFD);
  }
  for (int si = 0; si < 6; ++si) {
    long long o = (long long)si * HH * HH;
    LAUNCH_N(k_conv_bf16_mask, (long long)HH * HH, w_skip + o, SKIP_W + o, SKIP_MASKS + o, (long long)HH * HH);
  }
  LAUNCH_N(k_conv_bf16_mask, (long long)PQ * HH, w_head, HEAD_W, HEAD_MASK, (long long)PQ * HH);

  // ---- patch embedding ----
  LAUNCH_N(k_extract_patches, (long long)NTOK * PQ, xp_bf, X);
  {
    GP p = gp(xp_bf, PQ, w_patch, PQ, hbuf, HH, NTOK, HH, PQ);
    p.bias = PATCH_B;
    gemm_launch<0, 1, 0, 0, 0, 2, 4>(stream, p, 1);
  }
  LAUNCH_N(k_add_pos, (long long)NTOK * HH, hbuf, TPOS, SPOS);
  LAUNCH_N(k_conv_bf16, (long long)NTOK * HH, act_bf[0], hbuf, (long long)NTOK * HH);

  static const int SKIP_SRC[6] = {0, 0, 1, 0, 1, 2};
  static const int SKIP_TGT[6] = {2, 3, 3, 4, 4, 4};

  for (int l = 0; l < LL; ++l) {
    long long oHH = (long long)l * HH * HH;
    long long oFH = (long long)l * FFD * HH;

    // ================= spatial attention =================
    k_layernorm<<<NTOK, 256, 0, stream>>>(ln_bf, hbuf, LN_S_G + l * HH, LN_S_B + l * HH);
    { GP p = gp(ln_bf, HH, w_sq + oHH, HH, q_bf, HH, NTOK, HH, HH);
      p.bias = SQ_B + l * HH; gemm_launch<1, 1, 0, 0, 0, 2, 4>(stream, p, 1); }
    { GP p = gp(ln_bf, HH, w_sk + oHH, HH, k_bf, HH, NTOK, HH, HH);
      p.bias = SK_B + l * HH; gemm_launch<1, 1, 0, 0, 0, 2, 4>(stream, p, 1); }
    { GP p = gp(ln_bf, HH, w_sv + oHH, HH, vbuf, HH, NTOK, HH, HH);
      p.bias = SV_B + l * HH; gemm_launch<0, 1, 0, 0, 0, 2, 4>(stream, p, 1); }
    LAUNCH_N(k_vt_spatial, (long long)BT * HH * PP_, vt_bf, vbuf);
    { // scores: per (bt, head), 256x256, K=64; epilogue scale + mask bias, bf16 out
      GP p = gp(q_bf, HH, k_bf, HH, sc_bf, PP_, PP_, PP_, HD);
      p.c1 = NHD; p.c0 = 1;
      p.sA2 = (long long)PP_ * HH; p.sA1 = HD;
      p.sB2 = (long long)PP_ * HH; p.sB1 = HD;
      p.sC2 = (long long)NHD * PP_ * PP_; p.sC1 = (long long)PP_ * PP_;
      p.mask = S_MASK; p.mld = PP_; p.scale = 0.125f;
      gemm_launch<1, 0, 0, 0, 1, 2, 4>(stream, p, BT * NHD);
    }
    k_softmax256<<<(BT * NHD * PP_) / 4, 128, 0, stream>>>(sc_bf, (long long)BT * NHD * PP_);
    { // O = probs @ V : per (bt, head), 256x64, K=256, bf16 out into head slice
      GP p = gp(sc_bf, PP_, vt_bf, PP_, o_bf, HH, PP_, HD, PP_);
      p.c1 = NHD; p.c0 = 1;
      p.sA2 = (long long)NHD * PP_ * PP_; p.sA1 = (long long)PP_ * PP_;
      p.sB2 = (long long)HH * PP_;        p.sB1 = (long long)HD * PP_;
      p.sC2 = (long long)PP_ * HH;        p.sC1 = HD;
      gemm_launch<1, 0, 0, 0, 0, 2, 4>(stream, p, BT * NHD);
    }
    { GP p = gp(o_bf, HH, w_so + oHH, HH, hbuf, HH, NTOK, HH, HH);
      p.res = hbuf; gemm_launch<0, 0, 0, 1, 0, 2, 4>(stream, p, 1); }

    // ================= temporal attention =================
    k_layernorm<<<NTOK, 256, 0, stream>>>(ln_bf, hbuf, LN_T_G + l * HH, LN_T_B + l * HH);
    { GP p = gp(ln_bf, HH, w_tq + oHH, HH, q_bf, HH, NTOK, HH, HH);
      p.bias = TQ_B + l * HH; gemm_launch<1, 1, 0, 0, 0, 2, 4>(stream, p, 1); }
    { GP p = gp(ln_bf, HH, w_tk + oHH, HH, k_bf, HH, NTOK, HH, HH);
      p.bias = TK_B + l * HH; gemm_launch<1, 1, 0, 0, 0, 2, 4>(stream, p, 1); }
    { GP p = gp(ln_bf, HH, w_tv + oHH, HH, vbuf, HH, NTOK, HH, HH);
      p.bias = TV_B + l * HH; gemm_launch<0, 1, 0, 0, 0, 2, 4>(stream, p, 1); }
    LAUNCH_N(k_vt_temporal, (long long)BZ * PP_ * HH * 32, vt_bf, vbuf);
    { // scores: per (b, p, head), 16x16, K=64, stored with ldc=32 (K-pad for next GEMM)
      GP p = gp(q_bf, PP_ * HH, k_bf, PP_ * HH, sc_bf, 32, TT, TT, HD);
      p.c1 = PP_; p.c0 = NHD;
      p.sA2 = (long long)TT * PP_ * HH; p.sA1 = HH; p.sA0 = HD;
      p.sB2 = (long long)TT * PP_ * HH; p.sB1 = HH; p.sB0 = HD;
      p.sC2 = (long long)PP_ * NHD * TT * 32; p.sC1 = (long long)NHD * TT * 32; p.sC0 = (long long)TT * 32;
      p.mask = T_MASK; p.mld = TT; p.scale = 0.125f;
      gemm_launch<1, 0, 0, 0, 1, 1, 1>(stream, p, BZ * PP_ * NHD);
    }
    k_softmax16<<<(BZ * PP_ * NHD * TT + 255) / 256, 256, 0, stream>>>(
        sc_bf, (long long)BZ * PP_ * NHD * TT);
    { // O = probs @ V : per (b, p, head), 16x64, K=32 (zero-padded)
      GP p = gp(sc_bf, 32, vt_bf, 32, o_bf, PP_ * HH, TT, HD, 32);
      p.c1 = PP_; p.c0 = NHD;
      p.sA2 = (long long)PP_ * NHD * TT * 32; p.sA1 = (long long)NHD * TT * 32; p.sA0 = (long long)TT * 32;
      p.sB2 = (long long)PP_ * HH * 32;       p.sB1 = (long long)HH * 32;       p.sB0 = (long long)HD * 32;
      p.sC2 = (long long)TT * PP_ * HH;       p.sC1 = HH;                       p.sC0 = HD;
      gemm_launch<1, 0, 0, 0, 0, 1, 4>(stream, p, BZ * PP_ * NHD);
    }
    { GP p = gp(o_bf, HH, w_to + oHH, HH, hbuf, HH, NTOK, HH, HH);
      p.res = hbuf; gemm_launch<0, 0, 0, 1, 0, 2, 4>(stream, p, 1); }

    // ================= feed-forward =================
    k_layernorm<<<NTOK, 256, 0, stream>>>(ln_bf, hbuf, LN_F_G + l * HH, LN_F_B + l * HH);
    { GP p = gp(ln_bf, HH, w_ff1 + oFH, HH, ff1_bf, FFD, NTOK, FFD, HH);
      p.bias = FF1_B + (long long)l * FFD;
      gemm_launch<1, 1, 1, 0, 0, 2, 4>(stream, p, 1); }     // +bias, ReLU, bf16 out
    { GP p = gp(ff1_bf, FFD, w_ff2 + oFH, FFD, hbuf, HH, NTOK, HH, FFD);
      p.bias = FF2_B + (long long)l * HH; p.res = hbuf;
      gemm_launch<0, 1, 0, 1, 0, 2, 4>(stream, p, 1); }     // +bias, +residual

    // ================= skip connections into act l+1 =================
    for (int si = 0; si < 6; ++si) {
      if (SKIP_TGT[si] != l + 1) continue;
      GP p = gp(act_bf[SKIP_SRC[si]], HH, w_skip + (long long)si * HH * HH, HH,
                hbuf, HH, NTOK, HH, HH);
      p.res = hbuf;
      gemm_launch<0, 0, 0, 1, 0, 2, 4>(stream, p, 1);
    }
    if (l + 1 <= 2)
      LAUNCH_N(k_conv_bf16, (long long)NTOK * HH, act_bf[l + 1], hbuf, (long long)NTOK * HH);
  }

  // ================= head =================
  k_layernorm<<<NTOK, 256, 0, stream>>>(ln_bf, hbuf, LNF_G, LNF_B);
  { // last timestep only: rows (b, T-1, p); batched over b
    GP p = gp(ln_bf + (long long)(TT - 1) * PP_ * HH, HH, w_head, HH,
              patches, PQ, PP_, PQ, HH);
    p.bias = HEAD_B;
    p.sA2 = (long long)TT * PP_ * HH;
    p.sC2 = (long long)PP_ * PQ;
    gemm_launch<0, 1, 0, 0, 0, 2, 4>(stream, p, BZ);
  }
  LAUNCH_N(k_unpatch_sigmoid, (long long)BZ * IMGD * IMGD, (float*)d_out, patches);
}
